// MSDA_25598005084486
// MI455X (gfx1250) — matrix operand order repaired and checked by reading, not compiled, not run on hardware
//
#include <hip/hip_runtime.h>
#include <hip/hip_bf16.h>
#include <stdint.h>

// ---------- types ----------
typedef __attribute__((ext_vector_type(16))) __bf16        v16bf;
typedef __attribute__((ext_vector_type(8)))  float         v8f;
typedef __attribute__((ext_vector_type(4)))  unsigned int  u32x4;
typedef __attribute__((ext_vector_type(2)))  float         f32x2;

union FragAB { v16bf v; u32x4 q[2]; };

// ---------- problem constants ----------
#define HH     128
#define WW     128
#define CC     256
#define PIX    16384            // H*W per batch
#define BATCH  4
#define MROWS  65536            // BATCH*PIX  (global pixel rows)
#define KDIM   256
#define NQKV   768

// ---------- workspace layout (bytes, 256-aligned) ----------
#define WS_XT     ((size_t)0)            // bf16 Xt[MROWS][256]        32 MiB
#define WS_WQ     ((size_t)33554432)     // bf16 Wq[768][256]          384 KiB
#define WS_WP     ((size_t)33947648)     // bf16 Wp[256][256]          128 KiB
#define WS_STATS  ((size_t)34078720)     // f32 stats[512] (+pad)      4 KiB
#define WS_QKV    ((size_t)34082816)     // f32 QKV[MROWS][768]        192 MiB
#define WS_YT     ((size_t)235409408)    // bf16 Yt[MROWS][256]        32 MiB
#define WS_Z      ((size_t)268963840)    // f32 Z[MROWS][256]          64 MiB

// ---------------------------------------------------------------------------
// 1) Transpose x (b,c,p) f32 -> Xt (b*PIX+p, c) bf16, LDS-tiled 64x64.
// ---------------------------------------------------------------------------
__global__ __launch_bounds__(256)
void msda_transpose_x(const float* __restrict__ x, __bf16* __restrict__ Xt) {
    __shared__ float t[64 * 65];
    const int b     = blockIdx.z;
    const int cBase = blockIdx.y * 64;
    const int pBase = blockIdx.x * 64;
    #pragma unroll
    for (int it = 0; it < 16; ++it) {
        int flat = it * 256 + threadIdx.x;
        int ci = flat >> 6, pj = flat & 63;
        t[ci * 65 + pj] = x[((size_t)b * CC + cBase + ci) * PIX + pBase + pj];
    }
    __syncthreads();
    #pragma unroll
    for (int it = 0; it < 16; ++it) {
        int flat = it * 256 + threadIdx.x;
        int pj = flat >> 6, ci = flat & 63;
        Xt[((size_t)(b * PIX + pBase + pj)) * CC + cBase + ci] = (__bf16)t[ci * 65 + pj];
    }
}

// ---------------------------------------------------------------------------
// 2) Flat f32 -> bf16 convert (weights).
// ---------------------------------------------------------------------------
__global__ __launch_bounds__(256)
void msda_cvt(const float* __restrict__ s, __bf16* __restrict__ d, int n) {
    int i = blockIdx.x * 256 + threadIdx.x;
    if (i < n) d[i] = (__bf16)s[i];
}

// ---------------------------------------------------------------------------
// 3) GEMM: D[P][o] = sum_c A[P][c] * Bm[o][c]   (A,Bm bf16 K-innermost, D f32)
//    Per wave: 16(M) x 64(N) tile, K=256 in 8 steps of 32, 4 WMMAs/step.
//    A fragment (contiguous k order, shared with B): lane(m=lane&15, half=lane>>4):
//      vgpr j<4 : K = 16*half + 2j     -> contiguous dword4 at k0 + 16*half
//      vgpr j>=4: K = 16*half + 2j     -> contiguous dword4 at k0 + 16*half + 8
//    B fragment (same contiguous k order as A): lane(n=lane&15): row o=oBase+n,
//      vgpr j: K = 16*half + 2j       -> two contiguous dword4s.
// ---------------------------------------------------------------------------
__global__ __launch_bounds__(256)
void msda_gemm_bf16(const __bf16* __restrict__ A, const __bf16* __restrict__ Bm,
                    float* __restrict__ D, int Ntot) {
    const int lane  = threadIdx.x & 31;
    const int wave  = threadIdx.x >> 5;
    const int half  = lane >> 4;
    const int mr    = lane & 15;
    const int pBase = (blockIdx.x * 8 + wave) * 16;
    const int oBase = blockIdx.y * 64;

    const __bf16* rowA = A  + (size_t)(pBase + mr) * KDIM;
    const __bf16* rowB = Bm + (size_t)(oBase + mr) * KDIM + (half << 4);

    v8f acc0 = {}, acc1 = {}, acc2 = {}, acc3 = {};
    #pragma unroll
    for (int k0 = 0; k0 < KDIM; k0 += 32) {
        FragAB af;
        af.q[0] = *(const u32x4*)(rowA + k0 + half * 16);
        af.q[1] = *(const u32x4*)(rowA + k0 + half * 16 + 8);

        FragAB b0, b1, b2, b3;
        const __bf16* bp = rowB + k0;
        b0.q[0] = *(const u32x4*)(bp);                 b0.q[1] = *(const u32x4*)(bp + 8);
        b1.q[0] = *(const u32x4*)(bp + 16 * KDIM);     b1.q[1] = *(const u32x4*)(bp + 16 * KDIM + 8);
        b2.q[0] = *(const u32x4*)(bp + 32 * KDIM);     b2.q[1] = *(const u32x4*)(bp + 32 * KDIM + 8);
        b3.q[0] = *(const u32x4*)(bp + 48 * KDIM);     b3.q[1] = *(const u32x4*)(bp + 48 * KDIM + 8);

        acc0 = __builtin_amdgcn_wmma_f32_16x16x32_bf16(false, af.v, false, b0.v, (short)0, acc0, false, false);
        acc1 = __builtin_amdgcn_wmma_f32_16x16x32_bf16(false, af.v, false, b1.v, (short)0, acc1, false, false);
        acc2 = __builtin_amdgcn_wmma_f32_16x16x32_bf16(false, af.v, false, b2.v, (short)0, acc2, false, false);
        acc3 = __builtin_amdgcn_wmma_f32_16x16x32_bf16(false, af.v, false, b3.v, (short)0, acc3, false, false);
    }
    // C/D layout: vgpr i -> M = i + 8*half, N = lane&15
    #pragma unroll
    for (int i = 0; i < 8; ++i) {
        size_t base = (size_t)(pBase + i + half * 8) * Ntot + oBase + mr;
        D[base]      = acc0[i];
        D[base + 16] = acc1[i];
        D[base + 32] = acc2[i];
        D[base + 48] = acc3[i];
    }
}

// ---------------------------------------------------------------------------
// 4) Dilated neighborhood attention. One wave per (pixel, head).
//    Lane L owns d = {2L, 2L+1}. OOB neighbors: score = 0 (kept in softmax),
//    value = 0 (skipped in the weighted sum). Writes Yt bf16 pixel-major.
// ---------------------------------------------------------------------------
__global__ __launch_bounds__(256)
void msda_attn(const float* __restrict__ QKV, __bf16* __restrict__ Yt) {
    const int lane = threadIdx.x & 31;
    const int wave = threadIdx.x >> 5;
    const int pairIdx = blockIdx.x * 8 + wave;       // < 262144
    const int hi = pairIdx & 3;
    const int P  = pairIdx >> 2;
    const int b  = P >> 14;
    const int p  = P & (PIX - 1);
    const int yy = p >> 7;
    const int xx = p & (WW - 1);
    const int r  = hi + 1;
    const int qc = hi * 64 + 2 * lane;

    f32x2 q = *(const f32x2*)(QKV + (size_t)P * NQKV + qc);

    float sc[9];
    int   nbP[9];
    #pragma unroll
    for (int i = 0; i < 3; ++i) {
        #pragma unroll
        for (int j = 0; j < 3; ++j) {
            const int n  = i * 3 + j;
            const int py = yy + (i - 1) * r;
            const int px = xx + (j - 1) * r;
            const bool ok = (py >= 0) & (py < HH) & (px >= 0) & (px < WW);
            const int Pn = (b << 14) + (py << 7) + px;
            nbP[n] = ok ? Pn : -1;
            float s = 0.f;
            if (ok) {
                f32x2 k2 = *(const f32x2*)(QKV + (size_t)Pn * NQKV + 256 + qc);
                s = q.x * k2.x + q.y * k2.y;
            }
            #pragma unroll
            for (int off = 16; off > 0; off >>= 1) s += __shfl_xor(s, off, 32);
            sc[n] = ok ? s : 0.f;
        }
    }
    float mx = sc[0];
    #pragma unroll
    for (int n = 1; n < 9; ++n) mx = fmaxf(mx, sc[n]);
    float es[9], sum = 0.f;
    #pragma unroll
    for (int n = 0; n < 9; ++n) { es[n] = __expf(sc[n] - mx); sum += es[n]; }
    const float inv = 1.f / sum;

    float o0 = 0.f, o1 = 0.f;
    #pragma unroll
    for (int n = 0; n < 9; ++n) {
        if (nbP[n] >= 0) {
            f32x2 v2 = *(const f32x2*)(QKV + (size_t)nbP[n] * NQKV + 512 + qc);
            const float w = es[n] * inv;
            o0 += w * v2.x;
            o1 += w * v2.y;
        }
    }
    union { __bf16 h[2]; unsigned int u; } pk;
    pk.h[0] = (__bf16)o0;
    pk.h[1] = (__bf16)o1;
    *(unsigned int*)(Yt + (size_t)P * CC + qc) = pk.u;
}

// ---------------------------------------------------------------------------
// 5) BN statistics: per-channel sum & sumsq over 65536 rows.
// ---------------------------------------------------------------------------
__global__ __launch_bounds__(256)
void msda_zero(float* __restrict__ p, int n) {
    int i = blockIdx.x * 256 + threadIdx.x;
    if (i < n) p[i] = 0.f;
}

__global__ __launch_bounds__(256)
void msda_bnstats(const float* __restrict__ Z, float* __restrict__ stats) {
    const int c = threadIdx.x;                // channel
    const int rows = MROWS / 128;             // gridDim.x == 128
    const int r0 = blockIdx.x * rows;
    float s = 0.f, sq = 0.f;
    for (int rw = r0; rw < r0 + rows; ++rw) {
        float v = Z[(size_t)rw * CC + c];     // coalesced across threads
        s += v;
        sq += v * v;
    }
    atomicAdd(&stats[c], s);
    atomicAdd(&stats[CC + c], sq);
}

// ---------------------------------------------------------------------------
// 6) Finalize: BN normalize + affine + SiLU + residual, with LDS transpose
//    (pad 257 -> conflict-free) so Z reads, x reads and out writes coalesce.
// ---------------------------------------------------------------------------
__global__ __launch_bounds__(256)
void msda_finalize(const float* __restrict__ Z, const float* __restrict__ x,
                   const float* __restrict__ stats, const float* __restrict__ gamma,
                   const float* __restrict__ beta, float* __restrict__ out) {
    __shared__ float t[32 * 257];
    const int b     = blockIdx.y;
    const int pBase = blockIdx.x * 32;
    #pragma unroll
    for (int it = 0; it < 32; ++it) {
        int flat = it * 256 + threadIdx.x;
        int pj = flat >> 8, ci = flat & 255;
        t[pj * 257 + ci] = Z[((size_t)(b * PIX + pBase + pj)) * CC + ci];
    }
    __syncthreads();
    const float invN = 1.f / (float)MROWS;
    #pragma unroll
    for (int it = 0; it < 32; ++it) {
        int flat = it * 256 + threadIdx.x;
        int c = flat >> 5, pj = flat & 31;
        float z    = t[pj * 257 + c];
        float mean = stats[c] * invN;
        float var  = fmaxf(stats[CC + c] * invN - mean * mean, 0.f);
        float yn   = (z - mean) * rsqrtf(var + 1e-5f) * gamma[c] + beta[c];
        float si   = yn / (1.f + __expf(-yn));
        size_t xi  = ((size_t)b * CC + c) * PIX + pBase + pj;
        out[xi] = x[xi] + si;
    }
}

// ---------------------------------------------------------------------------
extern "C" void kernel_launch(void* const* d_in, const int* in_sizes, int n_in,
                              void* d_out, int out_size, void* d_ws, size_t ws_size,
                              hipStream_t stream) {
    const float* x      = (const float*)d_in[0];
    const float* w_qkv  = (const float*)d_in[1];
    const float* w_proj = (const float*)d_in[2];
    const float* gamma  = (const float*)d_in[3];
    const float* beta   = (const float*)d_in[4];
    float* out = (float*)d_out;

    char* ws = (char*)d_ws;
    __bf16* Xt    = (__bf16*)(ws + WS_XT);
    __bf16* Wq    = (__bf16*)(ws + WS_WQ);
    __bf16* Wp    = (__bf16*)(ws + WS_WP);
    float*  stats = (float*) (ws + WS_STATS);
    float*  QKV   = (float*) (ws + WS_QKV);
    __bf16* Yt    = (__bf16*)(ws + WS_YT);
    float*  Z     = (float*) (ws + WS_Z);

    // x -> bf16 pixel-major; weights -> bf16
    msda_transpose_x<<<dim3(PIX / 64, CC / 64, BATCH), 256, 0, stream>>>(x, Xt);
    msda_cvt<<<(NQKV * CC + 255) / 256, 256, 0, stream>>>(w_qkv, Wq, NQKV * CC);
    msda_cvt<<<(CC * CC + 255) / 256, 256, 0, stream>>>(w_proj, Wp, CC * CC);

    // QKV projection: [65536 x 256] * [256 x 768]
    msda_gemm_bf16<<<dim3(MROWS / 128, NQKV / 64), 256, 0, stream>>>(Xt, Wq, QKV, NQKV);

    // dilated neighborhood attention (4 heads, rates 1..4)
    msda_attn<<<(MROWS * 4) / 8, 256, 0, stream>>>(QKV, Yt);

    // output projection: [65536 x 256] * [256 x 256]
    msda_gemm_bf16<<<dim3(MROWS / 128, CC / 64), 256, 0, stream>>>(Yt, Wp, Z, CC);

    // batch-norm stats + finalize (BN, SiLU, residual)
    msda_zero<<<2, 256, 0, stream>>>(stats, 2 * CC);
    msda_bnstats<<<128, 256, 0, stream>>>(Z, stats);
    msda_finalize<<<dim3(PIX / 32, BATCH), 256, 0, stream>>>(Z, x, stats, gamma, beta, out);
}
